// Encoder_GCN_18339510354226
// MI455X (gfx1250) — compile-verified, matrix operand order checked by reading, not run on hardware
//
#include <hip/hip_runtime.h>
#include <hip/hip_bf16.h>

typedef __attribute__((ext_vector_type(16))) _Float16     v16h;
typedef __attribute__((ext_vector_type(8)))  float        v8f;
typedef __attribute__((ext_vector_type(4)))  unsigned int v4u;

#define N_NODES 50000
#define HID 64
#define DOUT 64

// ---------------- generic fills ----------------
__global__ void k_fill_f32(float* p, float v, int n) {
    int i = blockIdx.x * blockDim.x + threadIdx.x;
    if (i < n) p[i] = v;
}
__global__ void k_fill_i32(int* p, int v, int n) {
    int i = blockIdx.x * blockDim.x + threadIdx.x;
    if (i < n) p[i] = v;
}

// ---------------- feature build (ordered: exits -> visited -> current) ----------------
__global__ void k_scatter_exits(float* feat, const int* exits, int nex, int B, int N) {
    int i = blockIdx.x * blockDim.x + threadIdx.x;
    if (i < B * nex) {
        int b = i / nex;
        feat[(size_t)b * N + exits[i % nex]] = 1.0f;
    }
}
__global__ void k_scatter_hist(float* feat, const int* hist, int B, int H, int N) {
    int i = blockIdx.x * blockDim.x + threadIdx.x;
    int tot = B * (H - 1);
    if (i < tot) {
        int b = i / (H - 1), j = i % (H - 1);
        feat[(size_t)b * N + hist[b * H + j]] = 0.1f;
    }
}
__global__ void k_scatter_cur(float* feat, const int* hist, int B, int H, int N) {
    int i = blockIdx.x * blockDim.x + threadIdx.x;
    if (i < B) feat[(size_t)i * N + hist[i * H + H - 1]] = 0.5f;
}

// ---------------- CSR build (counting sort by dst) ----------------
__global__ void k_count(const int* dst, int* deg, int E) {
    int i = blockIdx.x * blockDim.x + threadIdx.x;
    if (i < E) atomicAdd(&deg[dst[i]], 1);
}
__global__ void k_scan_chunk(const int* deg, int* incl, int* csum, int N) {
    __shared__ int lds[512];
    int t = threadIdx.x;
    int idx = blockIdx.x * 512 + t;
    int v = (idx < N) ? deg[idx] : 0;
    lds[t] = v;
    __syncthreads();
    for (int s = 1; s < 512; s <<= 1) {
        int x = (t >= s) ? lds[t - s] : 0;
        __syncthreads();
        lds[t] += x;
        __syncthreads();
    }
    if (idx < N) incl[idx] = lds[t];
    if (t == 511) csum[blockIdx.x] = lds[511];
}
__global__ void k_scan_chunks(int* csum, int nch) {
    if (blockIdx.x == 0 && threadIdx.x == 0) {
        int off = 0;
        for (int i = 0; i < nch; i++) { int x = csum[i]; csum[i] = off; off += x; }
        csum[nch] = off;
    }
}
__global__ void k_rowptr(const int* incl, const int* csum, int* rp, int N) {
    int i = blockIdx.x * blockDim.x + threadIdx.x;
    if (i < N) {
        rp[i + 1] = incl[i] + csum[i / 512];
        if (i == 0) rp[0] = 0;
    }
}
__global__ void k_bucket(const int* src, const int* dst, const int* rp, int* fillc,
                         int* srcs, int E) {
    int i = blockIdx.x * blockDim.x + threadIdx.x;
    if (i < E) {
        int d = dst[i];
        int p = rp[d] + atomicAdd(&fillc[d], 1);
        srcs[p] = src[i];
    }
}

// ---------------- constants: cl=W1.al1, cr=W1.ar1, wl=W2@al2, wr=W2@ar2,
//                  W2 -> f16 pre-swizzled into WMMA B-fragment order ----------------
// w2pack layout: halves indexed by ((ft*32 + lane)*16 + h), ft = nt*2 + kstep.
// Element (K, ncol): K = ks*32 + (lane>=16?8:0) + 2*(h>>1) + ((h>>1)>=4?8:0) + (h&1),
//                    ncol = nt*16 + (lane&15).  => each fragment = 2 contiguous b128.
__global__ void k_consts(const float* W1, const float* al1, const float* ar1,
                         const float* W2, const float* al2, const float* ar2,
                         float* consts, float* wl, float* wr, _Float16* w2pack) {
    __shared__ float pl[64], pr[64];
    int k = threadIdx.x;   // 64 threads
    pl[k] = W1[k] * al1[k];
    pr[k] = W1[k] * ar1[k];
    float sl = 0.f, sr = 0.f;
    for (int d = 0; d < 64; d++) {
        float w = W2[k * 64 + d];
        sl += w * al2[d];
        sr += w * ar2[d];
    }
    wl[k] = sl; wr[k] = sr;
    // prepack B fragments (4096 halves total, 64 per thread)
    for (int i = 0; i < 64; ++i) {
        int g = k * 64 + i;
        int h    = g & 15;
        int lane = (g >> 4) & 31;
        int ft   = g >> 9;            // 0..7
        int nt = ft >> 1, ks = ft & 1;
        int v = h >> 1;
        int K = ks * 32 + ((lane >= 16) ? 8 : 0) + 2 * v + ((v >= 4) ? 8 : 0) + (h & 1);
        int ncol = nt * 16 + (lane & 15);
        w2pack[g] = (_Float16)W2[K * 64 + ncol];
    }
    __syncthreads();
    if (k == 0) {
        float a = 0.f, b = 0.f;
        for (int i = 0; i < 64; i++) { a += pl[i]; b += pr[i]; }
        consts[0] = a; consts[1] = b;
    }
}

// ---------------- layer 1: per-(b,node) online softmax over in-edges (scalar) ----------------
__global__ void k_layer1(const float* feat, const float* consts, const int* rp,
                         const int* srcs, float* s1, int B, int N) {
    int t = blockIdx.x * blockDim.x + threadIdx.x;
    if (t >= B * N) return;
    int b = t / N, n = t - b * N;
    float cl = consts[0], cr = consts[1];
    float erd = feat[t] * cr;
    int rs = rp[n], re = rp[n + 1];
    float m = -3.0e38f, den = 0.f, acc = 0.f;
    for (int j = rs; j < re; j++) {
        int s = srcs[j];
        float fs = feat[(size_t)b * N + s];
        float e = fs * cl + erd;
        e = (e >= 0.f) ? e : 0.2f * e;                 // leaky_relu 0.2
        float nm = fmaxf(m, e);
        float sc = __expf(m - nm), w = __expf(e - nm);
        den = den * sc + w;
        acc = acc * sc + w * fs;
        m = nm;
    }
    s1[t] = (re > rs) ? acc / den : 0.f;
}

// ---------------- h1 = relu(s1*W1 + b1) -> f16; el2/er2 = h1.(W2@al2), h1.(W2@ar2) ----------------
__global__ void k_h1(const float* s1, const float* W1, const float* b1,
                     const float* wl, const float* wr,
                     _Float16* h1h, float* el2, float* er2, int BN) {
    __shared__ float sl[4][64], sr[4][64];
    int d = threadIdx.x, g = threadIdx.y;
    int bn = blockIdx.x * 4 + g;
    float hv = 0.f;
    if (bn < BN) {
        hv = fmaxf(s1[bn] * W1[d] + b1[d], 0.f);
        h1h[(size_t)bn * 64 + d] = (_Float16)hv;
    }
    sl[g][d] = hv * wl[d];
    sr[g][d] = hv * wr[d];
    __syncthreads();
    for (int s = 32; s > 0; s >>= 1) {
        if (d < s) { sl[g][d] += sl[g][d + s]; sr[g][d] += sr[g][d + s]; }
        __syncthreads();
    }
    if (d == 0 && bn < BN) { el2[bn] = sl[g][0]; er2[bn] = sr[g][0]; }
}

// ---------------- WMMA GEMM: h2pre[BN,64] = h1h[BN,64] @ W2[64,64], f32 accum ----------------
// Wave computes 16 rows x 64 cols (4 N-tiles, 8 WMMAs); A loaded once as 4x b128.
// A row-major fragment mapping: u[0]=row dwords[4*hi .. 4*hi+3], u[1]=dwords[4*hi+8 .. +11];
// second k-step at +16 dwords.  B read from pre-swizzled w2pack, 2x b128 per fragment.
union Frag { v16h h; v4u u[2]; };

__global__ void __launch_bounds__(256) k_gemm(const _Float16* __restrict__ A,
                                              const _Float16* __restrict__ Bp,
                                              float* __restrict__ Cg, int BN) {
    int wave = threadIdx.x >> 5;
    int lane = threadIdx.x & 31;
    int m0 = blockIdx.x * 128 + wave * 16;
    int lr = lane & 15;
    int hi = lane >> 4;
    int row = m0 + lr;

    const v4u* ar = (const v4u*)(const void*)(A + (size_t)row * 64);  // 8 x b128 per row
    Frag a0, a1;
    a0.u[0] = ar[hi];     a0.u[1] = ar[hi + 2];
    a1.u[0] = ar[hi + 4]; a1.u[1] = ar[hi + 6];

    const v4u* bp = (const v4u*)(const void*)Bp;   // 2 x b128 per (ft,lane)
#pragma unroll
    for (int nt = 0; nt < 4; ++nt) {
        Frag b0, b1;
        int f0 = ((nt * 2 + 0) * 32 + lane) * 2;
        int f1 = ((nt * 2 + 1) * 32 + lane) * 2;
        b0.u[0] = bp[f0]; b0.u[1] = bp[f0 + 1];
        b1.u[0] = bp[f1]; b1.u[1] = bp[f1 + 1];
        v8f c = {};
        c = __builtin_amdgcn_wmma_f32_16x16x32_f16(false, a0.h, false, b0.h, (short)0, c,
                                                   false, false);
        c = __builtin_amdgcn_wmma_f32_16x16x32_f16(false, a1.h, false, b1.h, (short)0, c,
                                                   false, false);
#pragma unroll
        for (int r = 0; r < 8; ++r) {
            int orow = m0 + r + (hi ? 8 : 0);
            Cg[(size_t)orow * 64 + nt * 16 + lr] = c[r];
        }
    }
    (void)BN;
}

// ---------------- layer 2: one wave32 per (b,dst); online softmax + 64-vec accum;
//                  fused mean via LDS + 64 global atomics / block ----------------
__global__ void __launch_bounds__(256) k_agg(const float* __restrict__ el2,
                                             const float* __restrict__ er2,
                                             const float* __restrict__ h2,
                                             const float* __restrict__ b2,
                                             const int* __restrict__ rp,
                                             const int* __restrict__ srcs,
                                             float* __restrict__ out, int N, float invN) {
    __shared__ float bacc[64];
    int tx = threadIdx.x;
    if (tx < 64) bacc[tx] = 0.f;
    __syncthreads();
    int b = blockIdx.y;
    int n = blockIdx.x * 8 + (tx >> 5);
    int lane = tx & 31;
    size_t bn = (size_t)b * N + n;
    float erd = er2[bn];
    int rs = rp[n], re = rp[n + 1];
    float m = -3.0e38f, den = 0.f, a0 = 0.f, a1 = 0.f;
    for (int j = rs; j < re; j++) {
        int s = srcs[j];
        size_t bs = (size_t)b * N + s;
        float e = el2[bs] + erd;
        e = (e >= 0.f) ? e : 0.2f * e;
        float nm = fmaxf(m, e);
        float sc = __expf(m - nm), w = __expf(e - nm);
        den = den * sc + w;
        const float* hp = h2 + bs * 64;
        a0 = a0 * sc + w * hp[lane];
        a1 = a1 * sc + w * hp[lane + 32];
        m = nm;
    }
    float inv = (re > rs) ? 1.f / den : 0.f;
    float o0 = fmaxf(a0 * inv + b2[lane], 0.f);
    float o1 = fmaxf(a1 * inv + b2[lane + 32], 0.f);
    atomicAdd(&bacc[lane], o0);
    atomicAdd(&bacc[lane + 32], o1);
    __syncthreads();
    if (tx < 64) atomicAdd(&out[b * 64 + tx], bacc[tx] * invN);
}

extern "C" void kernel_launch(void* const* d_in, const int* in_sizes, int n_in,
                              void* d_out, int out_size, void* d_ws, size_t ws_size,
                              hipStream_t stream) {
    const int* hist  = (const int*)d_in[0];
    const int* exits = (const int*)d_in[1];
    const int* src   = (const int*)d_in[2];
    const int* dst   = (const int*)d_in[3];
    const float* W1  = (const float*)d_in[4];
    const float* al1 = (const float*)d_in[5];
    const float* ar1 = (const float*)d_in[6];
    const float* b1  = (const float*)d_in[7];
    const float* W2  = (const float*)d_in[8];
    const float* al2 = (const float*)d_in[9];
    const float* ar2 = (const float*)d_in[10];
    const float* b2  = (const float*)d_in[11];
    float* out = (float*)d_out;

    const int N = N_NODES;
    const int E = in_sizes[2];
    const int H = 50;
    const int B = in_sizes[0] / H;       // 8
    const int NEX = in_sizes[1];
    const int BN = B * N;                // 400000
    const int NCH = (N + 511) / 512;     // 98

    // workspace carve-out (~163 MB)
    char* base = (char*)d_ws;
    size_t off = 0;
    auto alloc = [&](size_t bytes) -> void* {
        void* p = base + off;
        off = (off + bytes + 255) & ~(size_t)255;
        return p;
    };
    float*    feat   = (float*)alloc((size_t)BN * 4);
    float*    s1     = (float*)alloc((size_t)BN * 4);
    float*    el2    = (float*)alloc((size_t)BN * 4);
    float*    er2    = (float*)alloc((size_t)BN * 4);
    _Float16* h1h    = (_Float16*)alloc((size_t)BN * HID * 2);
    float*    h2pre  = (float*)alloc((size_t)BN * DOUT * 4);
    int*      deg    = (int*)alloc((size_t)N * 4);
    int*      incl   = (int*)alloc((size_t)N * 4);
    int*      csum   = (int*)alloc((size_t)(NCH + 1) * 4);
    int*      rp     = (int*)alloc((size_t)(N + 1) * 4);
    int*      fillc  = (int*)alloc((size_t)N * 4);
    int*      srcs   = (int*)alloc((size_t)E * 4);
    float*    consts = (float*)alloc(2 * 4);
    float*    wl     = (float*)alloc(64 * 4);
    float*    wr     = (float*)alloc(64 * 4);
    _Float16* w2pack = (_Float16*)alloc(64 * 64 * 2);
    (void)ws_size; (void)n_in;

    // init
    k_fill_f32<<<(BN + 255) / 256, 256, 0, stream>>>(feat, 0.f, BN);
    k_fill_i32<<<(N + 255) / 256, 256, 0, stream>>>(deg, 0, N);
    k_fill_i32<<<(N + 255) / 256, 256, 0, stream>>>(fillc, 0, N);
    k_fill_f32<<<(out_size + 255) / 256, 256, 0, stream>>>(out, 0.f, out_size);

    // features (ordered writes: exits, visited, current)
    k_scatter_exits<<<(B * NEX + 255) / 256, 256, 0, stream>>>(feat, exits, NEX, B, N);
    k_scatter_hist<<<(B * (H - 1) + 255) / 256, 256, 0, stream>>>(feat, hist, B, H, N);
    k_scatter_cur<<<1, 256, 0, stream>>>(feat, hist, B, H, N);

    // CSR by dst
    k_count<<<(E + 255) / 256, 256, 0, stream>>>(dst, deg, E);
    k_scan_chunk<<<NCH, 512, 0, stream>>>(deg, incl, csum, N);
    k_scan_chunks<<<1, 32, 0, stream>>>(csum, NCH);
    k_rowptr<<<(N + 255) / 256, 256, 0, stream>>>(incl, csum, rp, N);
    k_bucket<<<(E + 255) / 256, 256, 0, stream>>>(src, dst, rp, fillc, srcs, E);

    // constants + B-fragment prepack
    k_consts<<<1, 64, 0, stream>>>(W1, al1, ar1, W2, al2, ar2, consts, wl, wr, w2pack);

    // layer 1 (scalar GAT), h1 + folded attention logits for layer 2
    k_layer1<<<(BN + 255) / 256, 256, 0, stream>>>(feat, consts, rp, srcs, s1, B, N);
    k_h1<<<BN / 4, dim3(64, 4), 0, stream>>>(s1, W1, b1, wl, wr, h1h, el2, er2, BN);

    // layer 2 GEMM via v_wmma_f32_16x16x32_f16 (8 WMMAs / wave, b128 fragment loads)
    k_gemm<<<BN / 128, 256, 0, stream>>>(h1h, w2pack, h2pre, BN);

    // layer 2 aggregation + fused mean
    k_agg<<<dim3(N / 8, B), 256, 0, stream>>>(el2, er2, h2pre, b2, rp, srcs, out, N,
                                              1.0f / (float)N);
}
